// SAMF_38285338476856
// MI455X (gfx1250) — compile-verified, matrix operand order checked by reading, not run on hardware
//
#include <hip/hip_runtime.h>
#include <math.h>
#include <stdint.h>

typedef __attribute__((ext_vector_type(16))) _Float16 v16h;
typedef __attribute__((ext_vector_type(8)))  float    v8f;
typedef __attribute__((ext_vector_type(4)))  unsigned int u32x4;
typedef __attribute__((ext_vector_type(8)))  int      i32x8;
typedef __attribute__((ext_vector_type(4)))  int      i32x4;

#define DEV __device__ __forceinline__

// ---------------------------------------------------------------------------
// WMMA fragment helpers (CDNA5 16x16x32 f16, wave32).
// ISA 7.12.2: lane&15 -> row (A) / col (B); half = lane>>4. The 16 packed
// f16 elements cover K as two contiguous 8-float runs: [8h,8h+8) and
// [16+8h,16+8h+8). So a unit-stride fragment = 4 x b128 loads.
// ---------------------------------------------------------------------------
DEV v16h frag_contig(const float* rowPtr, int half) {
  const float4* p0 = (const float4*)(rowPtr + (half << 3));
  const float4* p1 = (const float4*)(rowPtr + 16 + (half << 3));
  float4 q0 = p0[0], q1 = p0[1], q2 = p1[0], q3 = p1[1];
  v16h f;
  f[0] = (_Float16)q0.x;  f[1] = (_Float16)q0.y;
  f[2] = (_Float16)q0.z;  f[3] = (_Float16)q0.w;
  f[4] = (_Float16)q1.x;  f[5] = (_Float16)q1.y;
  f[6] = (_Float16)q1.z;  f[7] = (_Float16)q1.w;
  f[8] = (_Float16)q2.x;  f[9] = (_Float16)q2.y;
  f[10] = (_Float16)q2.z; f[11] = (_Float16)q2.w;
  f[12] = (_Float16)q3.x; f[13] = (_Float16)q3.y;
  f[14] = (_Float16)q3.z; f[15] = (_Float16)q3.w;
  return f;
}

// Strided (transposed-operand) gather fallback.
DEV v16h frag_strided(const float* rowPtr, long kStride, int half) {
  v16h f;
#pragma unroll
  for (int e = 0; e < 16; ++e) {
    int k = ((e & 8) ? 16 : 0) + (half << 3) + (e & 7);
    f[e] = (_Float16)rowPtr[(long)k * kStride];
  }
  return f;
}

// ---------------------------------------------------------------------------
// Tensor Data Mover: 2-D tile load Global->LDS (D# per ISA ch.8).
// data_size=4B; LDS padding via pad_interval=32 dwords / pad_amount=4 dwords
// -> padded LDS row stride of 36 floats (conflict-free for b128 reads).
// ---------------------------------------------------------------------------
DEV void tdm_load_tile_2d(unsigned lds_byte_off, const float* gptr,
                          unsigned tile_cols, unsigned tile_rows,
                          unsigned row_stride_elems) {
  unsigned long long ga = (unsigned long long)(uintptr_t)gptr;
  u32x4 g0;
  g0[0] = 1u;                                     // count=1, user descriptor
  g0[1] = lds_byte_off;                           // lds_addr (bytes)
  g0[2] = (unsigned)ga;                           // global_addr[31:0]
  g0[3] = (unsigned)((ga >> 32) & 0x01FFFFFFu)    // global_addr[56:32]
        | (2u << 30);                             // type = 2 ("image")
  i32x8 g1;
  g1[0] = (2 << 16)                               // data_size = 4 bytes
        | (1 << 20)                               // pad_enable
        | (4 << 22)                               // pad_interval: every 32 dw
        | (3 << 25);                              // pad_amount: 4 dwords
  g1[1] = (int)(0xFFFFu << 16);                   // tensor_dim0[15:0] (huge)
  g1[2] = 0x7FFF | (int)(0xFFFFu << 16);          // dim0[31:16] | dim1[15:0]
  g1[3] = 0x7FFF | (int)(tile_cols << 16);        // dim1[31:16] | tile_dim0
  g1[4] = (int)tile_rows;                         // tile_dim1 (tile_dim2 = 0)
  g1[5] = (int)row_stride_elems;                  // tensor_dim0_stride[31:0]
  g1[6] = 0;
  g1[7] = 0;
  i32x4 gz = {0, 0, 0, 0};
#if __has_include(<hip/amd_detail/amd_gfx1250_TDM.h>)
  i32x8 gz8 = {0, 0, 0, 0, 0, 0, 0, 0};
  __builtin_amdgcn_tensor_load_to_lds(g0, g1, gz, gz, gz8, 0);
#else
  __builtin_amdgcn_tensor_load_to_lds(g0, g1, gz, gz, 0);
#endif
}

// ---------------------------------------------------------------------------
// Generic small GEMM: D[32,N] = A[32,K] * B^T (+bias).
// BCONTIG: B element (n,k) at B[n*ldbN + k] (unit k-stride, vectorized);
// else at B[n*ldbN + k*ldbK] (strided gather). Wave per 16x16 tile.
// ---------------------------------------------------------------------------
template <bool BCONTIG>
__global__ void wmma_gemm_atb(const float* __restrict__ A, int lda,
                              const float* __restrict__ B, long ldbN, long ldbK,
                              const float* __restrict__ bias,
                              float* __restrict__ D, int ldd,
                              int N, int K) {
  int wid = blockIdx.x * (blockDim.x >> 5) + (threadIdx.x >> 5);
  int tilesN = N >> 4;
  if (wid >= 2 * tilesN) return;          // wave-uniform guard: EXEC stays full
  int tm = wid & 1, tn = wid >> 1;
  int lane = threadIdx.x & 31;
  int half = lane >> 4;
  const float* aRow = A + (long)(tm * 16 + (lane & 15)) * lda;
  const float* bRow = B + (long)(tn * 16 + (lane & 15)) * ldbN;
  v8f acc = {};
  for (int k = 0; k < K; k += 32) {
    v16h af = frag_contig(aRow + k, half);
    v16h bf = BCONTIG ? frag_contig(bRow + k, half)
                      : frag_strided(bRow + (long)k * ldbK, ldbK, half);
    acc = __builtin_amdgcn_wmma_f32_16x16x32_f16(false, af, false, bf,
                                                 (short)0, acc, false, false);
  }
  int n = tn * 16 + (lane & 15);
  float bv = bias ? bias[n] : 0.f;
#pragma unroll
  for (int r = 0; r < 8; ++r) {
    int m = tm * 16 + r + (half << 3);    // C/D layout: vgpr r -> M=r / M=r+8
    D[(long)m * ldd + n] = acc[r] + bv;
  }
}

// ---------------------------------------------------------------------------
// cbias[b] = sum_p b2d[p] * f3p[b,p]
// ---------------------------------------------------------------------------
__global__ void cbias_kernel(const float* __restrict__ fusedin,
                             const float* __restrict__ b2d,
                             float* __restrict__ cb) {
  int b = threadIdx.x >> 5;
  int lane = threadIdx.x & 31;
  const float* f3 = fusedin + (long)b * 9216;
  float acc = 0.f;
  for (int p = lane; p < 3072; p += 32) acc += b2d[p] * f3[p];
  for (int off = 16; off; off >>= 1) acc += __shfl_down(acc, off, 32);
  if (lane == 0) cb[b] = acc;
}

// ---------------------------------------------------------------------------
// scores_s[b,s] = feat_2d[b,s,:] . v[b,:] + cbias[b]   (wave per row)
// ---------------------------------------------------------------------------
__global__ void scores_s_kernel(const float* __restrict__ feat2d,
                                const float* __restrict__ v,
                                const float* __restrict__ cb,
                                float* __restrict__ ss) {
  int wid = blockIdx.x * (blockDim.x >> 5) + (threadIdx.x >> 5);
  if (wid >= 32 * 256) return;
  int b = wid >> 8;
  int lane = threadIdx.x & 31;
  const float* row = feat2d + (long)wid * 768;
  const float* vb  = v + b * 768;
  float acc = 0.f;
  for (int d = lane; d < 768; d += 32) acc += row[d] * vb[d];
  for (int off = 16; off; off >>= 1) acc += __shfl_down(acc, off, 32);
  if (lane == 0) ss[wid] = acc + cb[b];
}

// ---------------------------------------------------------------------------
// softmax over S=256 then wfeat[b,d] = sum_s w[s]*feat_2d[b,s,d]
// ---------------------------------------------------------------------------
__global__ void pool2d_stage1(const float* __restrict__ ss,
                              const float* __restrict__ feat2d,
                              float* __restrict__ wfeat) {
  __shared__ float w[256];
  __shared__ float red[256];
  int b = blockIdx.x, tid = threadIdx.x;
  float sc = ss[b * 256 + tid];
  red[tid] = sc; __syncthreads();
  for (int off = 128; off; off >>= 1) {
    if (tid < off) red[tid] = fmaxf(red[tid], red[tid + off]);
    __syncthreads();
  }
  float m = red[0]; __syncthreads();
  float e = __expf(sc - m);
  red[tid] = e; __syncthreads();
  for (int off = 128; off; off >>= 1) {
    if (tid < off) red[tid] += red[tid + off];
    __syncthreads();
  }
  w[tid] = e * (1.f / red[0]);
  __syncthreads();
  const float* f2b = feat2d + (long)b * 256 * 768;
  for (int d = tid; d < 768; d += 256) {
    float acc = 0.f;
    for (int s = 0; s < 256; ++s) acc += w[s] * f2b[(long)s * 768 + d];
    wfeat[b * 768 + d] = acc;
  }
}

// ---------------------------------------------------------------------------
// Text attention pooling: one pass over feat_text (flash-style split-T).
// ---------------------------------------------------------------------------
#define TCH 8
#define TROWS 64

__global__ void text_pass1(const float* __restrict__ ftext,
                           const float* __restrict__ fusedin,
                           float* __restrict__ pm, float* __restrict__ pl,
                           float* __restrict__ pacc) {
  __shared__ float f3s[3072];
  __shared__ float sc[TROWS];
  __shared__ float red[TROWS];
  int ch = blockIdx.x, b = blockIdx.y, tid = threadIdx.x;
  const float* f3p = fusedin + (long)b * 9216;
  for (int i = tid; i < 3072; i += 256) f3s[i] = f3p[i];
  __syncthreads();
  int w = tid >> 5, lane = tid & 31;
  const float* base = ftext + ((long)b * 512 + ch * TROWS) * 3072;
#pragma unroll
  for (int i = 0; i < TROWS / 8; ++i) {
    int tl = w + 8 * i;
    const float* row = base + (long)tl * 3072;
    float acc = 0.f;
    for (int p = lane; p < 3072; p += 32) acc += row[p] * f3s[p];
    for (int off = 16; off; off >>= 1) acc += __shfl_down(acc, off, 32);
    if (lane == 0) sc[tl] = acc;
  }
  __syncthreads();
  if (tid < TROWS) red[tid] = sc[tid];
  __syncthreads();
  for (int off = 32; off; off >>= 1) {
    if (tid < off) red[tid] = fmaxf(red[tid], red[tid + off]);
    __syncthreads();
  }
  float m = red[0];
  __syncthreads();
  if (tid < TROWS) { float e = __expf(sc[tid] - m); sc[tid] = e; red[tid] = e; }
  __syncthreads();
  for (int off = 32; off; off >>= 1) {
    if (tid < off) red[tid] += red[tid + off];
    __syncthreads();
  }
  if (tid == 0) { pm[b * TCH + ch] = m; pl[b * TCH + ch] = red[0]; }
  float acc[12];
#pragma unroll
  for (int j = 0; j < 12; ++j) acc[j] = 0.f;
  for (int t = 0; t < TROWS; ++t) {
    float wt = sc[t];
    const float* row = base + (long)t * 3072;
#pragma unroll
    for (int j = 0; j < 12; ++j) acc[j] += wt * row[tid + 256 * j];
  }
  float* pa = pacc + (long)(b * TCH + ch) * 3072;
#pragma unroll
  for (int j = 0; j < 12; ++j) pa[tid + 256 * j] = acc[j];
}

__global__ void text_pass2(const float* __restrict__ pm,
                           const float* __restrict__ pl,
                           const float* __restrict__ pacc,
                           float* __restrict__ fusedin) {
  int b = blockIdx.y;
  int p = blockIdx.x * 256 + threadIdx.x;
  float M = -1e30f;
  for (int c = 0; c < TCH; ++c) M = fmaxf(M, pm[b * TCH + c]);
  float L = 0.f, al[TCH];
  for (int c = 0; c < TCH; ++c) {
    al[c] = __expf(pm[b * TCH + c] - M);
    L += al[c] * pl[b * TCH + c];
  }
  float s = 0.f;
  for (int c = 0; c < TCH; ++c)
    s += al[c] * pacc[(long)(b * TCH + c) * 3072 + p];
  fusedin[(long)b * 9216 + 6144 + p] = s / L;
}

// ---------------------------------------------------------------------------
// Fused GEMM: [32,9216] x Wf[3072,9216]^T, split-K x8.
// TDM double-buffered LDS staging: wave 0 issues tensor_load_to_lds for the
// next k-step's A (32x32) and B (64x32) tiles while all 8 waves run WMMA on
// the current buffer; synchronized with s_wait_tensorcnt + barriers.
// TDM pad feature gives LDS row stride 36 floats (16B-aligned, conflict-free).
// ---------------------------------------------------------------------------
#define FK 9216
#define SPLITK 8
#define KC (FK / SPLITK)   // 1152
#define LSTRIDE 36

__global__ void fused_gemm_splitk(const float* __restrict__ fusedin,
                                  const float* __restrict__ Wf,
                                  float* __restrict__ part) {
  __shared__ __align__(16) float As[2][32 * LSTRIDE];
  __shared__ __align__(16) float Bs[2][64 * LSTRIDE];
  int nb = blockIdx.x, sk = blockIdx.y, tid = threadIdx.x;
  int w = tid >> 5, lane = tid & 31, half = lane >> 4;
  int tm = w & 1, tn = w >> 1;            // 2 x 4 subtiles of 16x16
  int nbase = nb * 64;
  int kb0 = sk * KC;
  const int steps = KC / 32;              // 36

  if (tid < 32) {                         // wave 0 preloads buffer 0
    tdm_load_tile_2d((unsigned)(uintptr_t)&As[0][0], fusedin + kb0, 32, 32, FK);
    tdm_load_tile_2d((unsigned)(uintptr_t)&Bs[0][0],
                     Wf + (long)nbase * FK + kb0, 32, 64, FK);
    __builtin_amdgcn_s_wait_tensorcnt(0);
  }
  __syncthreads();

  v8f acc = {};
  for (int i = 0; i < steps; ++i) {
    int cur = i & 1;
    if (i + 1 < steps && tid < 32) {      // overlap next DMA with compute
      int kb = kb0 + (i + 1) * 32;
      tdm_load_tile_2d((unsigned)(uintptr_t)&As[cur ^ 1][0],
                       fusedin + kb, 32, 32, FK);
      tdm_load_tile_2d((unsigned)(uintptr_t)&Bs[cur ^ 1][0],
                       Wf + (long)nbase * FK + kb, 32, 64, FK);
    }
    v16h af = frag_contig(&As[cur][(tm * 16 + (lane & 15)) * LSTRIDE], half);
    v16h bf = frag_contig(&Bs[cur][(tn * 16 + (lane & 15)) * LSTRIDE], half);
    acc = __builtin_amdgcn_wmma_f32_16x16x32_f16(false, af, false, bf,
                                                 (short)0, acc, false, false);
    __syncthreads();                      // all waves done reading buf[cur]
    if (tid < 32) __builtin_amdgcn_s_wait_tensorcnt(0);
    __syncthreads();                      // next buffer filled & visible
  }

  int n = nbase + tn * 16 + (lane & 15);
#pragma unroll
  for (int r = 0; r < 8; ++r) {
    int m = tm * 16 + r + (half << 3);
    part[((long)sk * 32 + m) * 3072 + n] = acc[r];
  }
}

__global__ void fused_reduce(const float* __restrict__ part,
                             const float* __restrict__ bfv,
                             float* __restrict__ fvec) {
  int idx = blockIdx.x * 256 + threadIdx.x;
  if (idx >= 32 * 3072) return;
  int q = idx % 3072, b = idx / 3072;
  float s = bfv[q];
  for (int c = 0; c < SPLITK; ++c) s += part[((long)c * 32 + b) * 3072 + q];
  fvec[idx] = s;
}

// out[b,s,p] = fvec[b,p]  (100 MB float4 stream)
__global__ void broadcast_out(const float4* __restrict__ fvec4,
                              float4* __restrict__ out4) {
  long idx = (long)blockIdx.x * 256 + threadIdx.x;
  const long total = 32L * 256 * 768;
  if (idx >= total) return;
  long b  = idx / (256 * 768);
  long p4 = idx % 768;
  out4[idx] = fvec4[b * 768 + p4];
}

// ---------------------------------------------------------------------------
extern "C" void kernel_launch(void* const* d_in, const int* in_sizes, int n_in,
                              void* d_out, int out_size, void* d_ws, size_t ws_size,
                              hipStream_t stream) {
  const float* feat3d = (const float*)d_in[0];
  const float* feat2d = (const float*)d_in[1];
  const float* ftext  = (const float*)d_in[2];
  const float* W3d    = (const float*)d_in[3];
  const float* b3d    = (const float*)d_in[4];
  const float* W2d    = (const float*)d_in[5];
  const float* b2d    = (const float*)d_in[6];
  const float* Wf     = (const float*)d_in[7];
  const float* bfv    = (const float*)d_in[8];

  float* ws      = (float*)d_ws;
  float* fusedin = ws;                       // [32,9216] concat(f3p|pooled2d|pooledtext)
  float* v       = fusedin + 32 * 9216;      // [32,768]
  float* cb      = v + 32 * 768;             // [32]
  float* ss      = cb + 32;                  // [32,256]
  float* wfeat   = ss + 32 * 256;            // [32,768]
  float* pm      = wfeat + 32 * 768;         // [32,8]
  float* pl      = pm + 256;                 // [32,8]
  float* pacc    = pl + 256;                 // [32*8,3072]
  float* part    = pacc;                     // aliased: lifetimes disjoint
  float* fvec    = pacc + 8 * 32 * 3072;     // [32,3072]

  // f3p = feat_3d @ W3d.T + b3d  -> fused_in[:, 0:3072]
  wmma_gemm_atb<true><<<dim3(96), dim3(128), 0, stream>>>(
      feat3d, 256, W3d, 256, 1, b3d, fusedin, 9216, 3072, 256);
  // v = f3p @ W2d   (B(n=d,k=p) = W2d[p*768+d], strided gather)
  wmma_gemm_atb<false><<<dim3(24), dim3(128), 0, stream>>>(
      fusedin, 9216, W2d, 1, 768, nullptr, v, 768, 768, 3072);
  cbias_kernel<<<dim3(1), dim3(1024), 0, stream>>>(fusedin, b2d, cb);
  scores_s_kernel<<<dim3(1024), dim3(256), 0, stream>>>(feat2d, v, cb, ss);
  pool2d_stage1<<<dim3(32), dim3(256), 0, stream>>>(ss, feat2d, wfeat);
  // pooled_2d = wfeat @ W2d.T + b2d -> fused_in[:, 3072:6144]
  wmma_gemm_atb<true><<<dim3(96), dim3(128), 0, stream>>>(
      wfeat, 768, W2d, 768, 1, b2d, fusedin + 3072, 9216, 3072, 768);
  text_pass1<<<dim3(TCH, 32), dim3(256), 0, stream>>>(ftext, fusedin, pm, pl, pacc);
  text_pass2<<<dim3(12, 32), dim3(256), 0, stream>>>(pm, pl, pacc, fusedin);
  fused_gemm_splitk<<<dim3(48, SPLITK), dim3(256), 0, stream>>>(fusedin, Wf, part);
  fused_reduce<<<dim3(384), dim3(256), 0, stream>>>(part, bfv, fvec);
  broadcast_out<<<dim3(24576), dim3(256), 0, stream>>>(
      (const float4*)fvec, (float4*)d_out);
}
